// TinyTransformer_70282844832120
// MI455X (gfx1250) — compile-verified, matrix operand order checked by reading, not run on hardware
//
#include <hip/hip_runtime.h>
#include <hip/hip_bf16.h>
#include <math.h>

// ---------------------------------------------------------------------------
// TinyTransformer forward on MI455X (gfx1250), wave32 + WMMA 16x16x32 f16.
// All operands kept as f16 shadows -> every fragment is a contiguous 128-bit
// load. Attention K-strips staged dense into LDS by the Tensor Data Mover.
// Both WMMA loops register-double-buffered to overlap loads with the XDL.
// ---------------------------------------------------------------------------

#define B_    32
#define S_    1024
#define D_    512
#define H_    16
#define NB_   12
#define DH_   32          // per-head dim (== WMMA K for f16)
#define TDH_  96          // 3*DH
#define FF_   2048
#define EPS_  1e-3f
#define SCALE_ 0.03125f   // 1/sqrt(S), folded into the f16 q copy
#define BS_   (B_ * S_)
#define FFN_CHUNK_ 4096
#define QROWS_ 32         // query rows per attention wave (2 M-tiles)

typedef __attribute__((ext_vector_type(16))) _Float16 v16h;
typedef __attribute__((ext_vector_type(8)))  _Float16 v8h;
typedef __attribute__((ext_vector_type(8)))  float    v8f;
typedef __attribute__((ext_vector_type(4)))  unsigned int v4u;
typedef __attribute__((ext_vector_type(8)))  int      v8i;
typedef __attribute__((ext_vector_type(4)))  int      v4i;

#if defined(__has_builtin)
#  if __has_builtin(__builtin_amdgcn_tensor_load_to_lds)
#    define HAVE_TDM 1
#  else
#    define HAVE_TDM 0
#  endif
#else
#  define HAVE_TDM 0
#endif

// ---------------- WMMA helpers (wave32, gfx1250 VGPR layouts) ---------------

__device__ __forceinline__ v8f wmma_f16(v16h a, v16h b, v8f c) {
  return __builtin_amdgcn_wmma_f32_16x16x32_f16(
      false, a, false, b, (short)0, c, false, false);
}

// Fragment from f16 row-major storage (two contiguous 128-bit loads).
// ISA 16-bit A layout: lane L<16 -> row M=L, elems 0..7 = K kb..kb+7,
// elems 8..15 = K kb+16..kb+23, kb=0; lanes 16..31 same rows, kb=8.
__device__ __forceinline__ v16h load_frag_f16(const _Float16* P, int ld, int row0, int k0) {
  int lane = threadIdx.x & 31;
  const _Float16* p = P + (size_t)(row0 + (lane & 15)) * ld + k0 + ((lane >> 4) << 3);
  v8h lo = *(const v8h*)p;
  v8h hi = *(const v8h*)(p + 16);
  v16h r;
#pragma unroll
  for (int e = 0; e < 8; ++e) { r[e] = lo[e]; r[8 + e] = hi[e]; }
  return r;
}

// ---------------- Weight prep: fp32 [K,N] -> f16 [N,K] ----------------------

__global__ void wt_convert_kernel(const float* __restrict__ W,
                                  _Float16* __restrict__ Wt, int K, int N) {
  int idx = blockIdx.x * blockDim.x + threadIdx.x;
  if (idx >= K * N) return;
  int k = idx / N, n = idx - k * N;
  Wt[(size_t)n * K + k] = (_Float16)W[idx];
}

// ---------------- Embedding + positional encoding (fp32 + f16 shadow) -------

__global__ void embed_pos_kernel(const int* __restrict__ tok,
                                 const float* __restrict__ emb,
                                 float* __restrict__ x,
                                 _Float16* __restrict__ x16) {
  int bs = blockIdx.x;
  int s  = bs & (S_ - 1);
  int t  = tok[bs];
  float* xo = x + (size_t)bs * D_;
  _Float16* ho = x16 + (size_t)bs * D_;
  const float* e = emb + (size_t)t * D_;
  for (int d = threadIdx.x; d < D_; d += blockDim.x) {
    float expnt = (2.0f * (float)(d >> 1)) / (float)D_;
    float ang   = (float)s / __powf(10000.0f, expnt);
    float pe    = (d & 1) ? __cosf(ang) : __sinf(ang);
    float v = e[d] + pe;
    xo[d] = v;
    ho[d] = (_Float16)v;
  }
}

// ---------------- WMMA GEMM: C = act(A16 * Wt^T + bias [+ Rsd]) -------------

template <int TN>
__global__ void gemm_wmma_kernel(const _Float16* __restrict__ A,
                                 const _Float16* __restrict__ Wt,
                                 const float* __restrict__ bias,
                                 const float* __restrict__ Rsd,
                                 float* __restrict__ C,
                                 _Float16* __restrict__ C16,
                                 _Float16* __restrict__ VT,
                                 int K, int lda, int ldc, int relu,
                                 int qcols, float qscale) {
  int n0   = blockIdx.x * (16 * TN);
  int row0 = blockIdx.y * 16;

  v8f acc[TN];
#pragma unroll
  for (int t = 0; t < TN; ++t) acc[t] = (v8f){};

#pragma unroll 2
  for (int k = 0; k < K; k += 32) {
    if (k + 32 < K)
      __builtin_prefetch(A + (size_t)row0 * lda + k + 32, 0, 1);  // global_prefetch_b8
    v16h a = load_frag_f16(A, lda, row0, k);
#pragma unroll
    for (int t = 0; t < TN; ++t) {
      v16h b = load_frag_f16(Wt, K, n0 + 16 * t, k);
      acc[t] = wmma_f16(a, b, acc[t]);
    }
  }

  int lane = threadIdx.x & 31;
  int rb   = row0 + ((lane >> 4) << 3);
  int nn   = lane & 15;
#pragma unroll
  for (int t = 0; t < TN; ++t) {
    int n = n0 + 16 * t + nn;
    float bv = bias ? bias[n] : 0.0f;
#pragma unroll
    for (int r = 0; r < 8; ++r) {
      int row = rb + r;
      float v = acc[t][r] + bv;
      if (Rsd)  v += Rsd[(size_t)row * ldc + n];
      if (relu) v = v > 0.0f ? v : 0.0f;
      if (C)    C[(size_t)row * ldc + n] = v;
      if (C16)  C16[(size_t)row * ldc + n] = (_Float16)((n < qcols) ? v * qscale : v);
      if (VT && n >= 2 * DH_)   // transposed V: VT[b][dim][key]
        VT[((size_t)(row >> 10) * DH_ + (n - 2 * DH_)) * S_ + (row & (S_ - 1))] = (_Float16)v;
    }
  }
}

// ---------------- Attention head (1 wave / 32 query rows) -------------------
// qkv16: [B,S,96] f16, q pre-scaled by 1/sqrt(S). K-strip (S x DH, stride 96)
// staged dense into LDS by the TDM. V read from VT[b][dim][key] (contiguous).
// LDS: 128 KB f32 scores (f16 probs aliased) + 64 KB K-strip = 192 KB / 320 KB.
// Each K tile feeds TWO independent WMMAs; P.V is a 2x2 accumulator block.

__global__ void attn_head_kernel(const _Float16* __restrict__ qkv16,
                                 const _Float16* __restrict__ vt,
                                 _Float16* __restrict__ hout16, int ld_h) {
  extern __shared__ char smem[];
  float*    sc   = (float*)smem;                        // QROWS_*1024 f32
  _Float16* p16  = (_Float16*)smem;                     // aliased probs (trailing writes)
  _Float16* klds = (_Float16*)(smem + QROWS_ * S_ * 4); // [1024][32] f16

  int b    = blockIdx.x / (S_ / QROWS_);
  int mt   = blockIdx.x % (S_ / QROWS_);
  int row0 = mt * QROWS_;
  const _Float16* base = qkv16 + (size_t)b * S_ * TDH_;
  const _Float16* Qh = base;
  const _Float16* Kh = base + DH_;
  const _Float16* Vt = vt + (size_t)b * DH_ * S_;

  int lane = threadIdx.x & 31;
  int rb   = (lane >> 4) << 3;
  int nn   = lane & 15;

  // ---- stage K strip into LDS via Tensor Data Mover ----
#if HAVE_TDM
  {
    unsigned lds_addr = __builtin_amdgcn_groupstaticsize() + (unsigned)(QROWS_ * S_ * 4);
    unsigned long long ga = (unsigned long long)(uintptr_t)Kh;
    v4u g0 = { 1u,                                      // count=1, user D#
               lds_addr,
               (unsigned)(ga & 0xFFFFFFFFu),
               (unsigned)(((ga >> 32) & 0x01FFFFFFu) | (2u << 30)) }; // hi | type=2
    v8i g1 = { (int)(1u << 16),                         // data_size=1 (2B)
               (int)((DH_ & 0xFFFF) << 16),             // tensor_dim0 lo16
               (int)((DH_ >> 16) | (S_ << 16)),         // dim0 hi | dim1 lo
               (int)((S_ >> 16) | (DH_ << 16)),         // dim1 hi | tile_dim0
               (int)S_,                                 // tile_dim1
               (int)TDH_,                               // dim0_stride lo32
               0, 0 };
    v4i gz = { 0, 0, 0, 0 };
#if __clang_major__ >= 23
    v8i gz8 = { 0, 0, 0, 0, 0, 0, 0, 0 };
    __builtin_amdgcn_tensor_load_to_lds(g0, g1, gz, gz, gz8, 0);
#else
    __builtin_amdgcn_tensor_load_to_lds(g0, g1, gz, gz, 0);
#endif
  }
#else
  for (int r = lane; r < S_; r += 32) {
    const v8h* src = (const v8h*)(Kh + (size_t)r * TDH_);
    v8h* dst = (v8h*)(klds + r * DH_);
    dst[0] = src[0]; dst[1] = src[1]; dst[2] = src[2]; dst[3] = src[3];
  }
#endif

  // overlap q fragment loads with the K staging
  v16h qa0 = load_frag_f16(Qh, TDH_, row0, 0);
  v16h qa1 = load_frag_f16(Qh, TDH_, row0 + 16, 0);

#if HAVE_TDM
  __builtin_amdgcn_s_wait_tensorcnt(0);
  asm volatile("" ::: "memory");
#else
  __syncthreads();
#endif

  // ---- pass 1: scores; double-buffered K tiles, 2 WMMAs per tile ----
  v16h kb_cur = load_frag_f16(klds, DH_, 0, 0);
  for (int kt = 0; kt < S_ / 16; ++kt) {
    v16h kb_nxt = load_frag_f16(klds, DH_, ((kt + 1) & (S_ / 16 - 1)) * 16, 0);
    v8f s0 = {}, s1 = {};
    s0 = wmma_f16(qa0, kb_cur, s0);
    s1 = wmma_f16(qa1, kb_cur, s1);
#pragma unroll
    for (int r = 0; r < 8; ++r) {
      sc[(rb + r) * S_ + kt * 16 + nn]        = s0[r];
      sc[(16 + rb + r) * S_ + kt * 16 + nn]   = s1[r];
    }
    kb_cur = kb_nxt;
  }
  __syncthreads();

  // ---- pass 2: softmax per row; normalized probs written f16 (aliased) ----
  for (int r = 0; r < QROWS_; ++r) {
    float* row = sc + r * S_;
    float m = -1e30f;
    for (int j = lane; j < S_; j += 32) m = fmaxf(m, row[j]);
#pragma unroll
    for (int off = 16; off > 0; off >>= 1) m = fmaxf(m, __shfl_xor(m, off));
    float sum = 0.0f;
    for (int j = lane; j < S_; j += 32) {
      float e = __expf(row[j] - m);
      row[j] = e;
      sum += e;
    }
#pragma unroll
    for (int off = 16; off > 0; off >>= 1) sum += __shfl_xor(sum, off);
    float inv = 1.0f / sum;
    // f16 write at byte 2*idx trails the f32 read at byte 4*idx -> safe alias
    for (int j = lane; j < S_; j += 32) p16[r * S_ + j] = (_Float16)(row[j] * inv);
  }
  __syncthreads();

  // ---- pass 3: out(32x32) = P(32x1024).V ; 2x2 accumulator block, ----------
  // ---- double-buffered P (LDS) and V^T (global) fragments            ----------
  v8f o0 = {}, o1 = {}, o2 = {}, o3 = {};
  v16h p0c = load_frag_f16(p16, S_, 0, 0);
  v16h p1c = load_frag_f16(p16, S_, 16, 0);
  v16h v0c = load_frag_f16(Vt, S_, 0, 0);
  v16h v1c = load_frag_f16(Vt, S_, 16, 0);
  for (int kc = 0; kc < S_; kc += 32) {
    int kn = (kc + 32) & (S_ - 1);
    v16h p0n = load_frag_f16(p16, S_, 0, kn);
    v16h p1n = load_frag_f16(p16, S_, 16, kn);
    v16h v0n = load_frag_f16(Vt, S_, 0, kn);
    v16h v1n = load_frag_f16(Vt, S_, 16, kn);
    o0 = wmma_f16(p0c, v0c, o0);
    o1 = wmma_f16(p0c, v1c, o1);
    o2 = wmma_f16(p1c, v0c, o2);
    o3 = wmma_f16(p1c, v1c, o3);
    p0c = p0n; p1c = p1n; v0c = v0n; v1c = v1n;
  }

  _Float16* out = hout16 + (size_t)(b * S_ + row0) * ld_h;
#pragma unroll
  for (int r = 0; r < 8; ++r) {
    out[(size_t)(rb + r) * ld_h + nn]           = (_Float16)o0[r];
    out[(size_t)(rb + r) * ld_h + 16 + nn]      = (_Float16)o1[r];
    out[(size_t)(16 + rb + r) * ld_h + nn]      = (_Float16)o2[r];
    out[(size_t)(16 + rb + r) * ld_h + 16 + nn] = (_Float16)o3[r];
  }
}

// ---------------- LayerNorm -> fp32 + f16 shadow ----------------------------

__global__ void ln_kernel(const float* __restrict__ X,
                          const float* __restrict__ g,
                          const float* __restrict__ be,
                          float* __restrict__ Y,
                          _Float16* __restrict__ Y16) {
  __shared__ float sm[8], sv[8];
  int row = blockIdx.x;
  const float* x = X + (size_t)row * D_;
  float s = 0.0f, s2 = 0.0f;
  for (int j = threadIdx.x; j < D_; j += blockDim.x) {
    float v = x[j];
    s += v; s2 += v * v;
  }
#pragma unroll
  for (int off = 16; off > 0; off >>= 1) {
    s  += __shfl_xor(s, off);
    s2 += __shfl_xor(s2, off);
  }
  int w = threadIdx.x >> 5, l = threadIdx.x & 31;
  if (l == 0) { sm[w] = s; sv[w] = s2; }
  __syncthreads();
  if (w == 0) {
    s  = (l < 8) ? sm[l] : 0.0f;
    s2 = (l < 8) ? sv[l] : 0.0f;
#pragma unroll
    for (int off = 4; off > 0; off >>= 1) {
      s  += __shfl_xor(s, off);
      s2 += __shfl_xor(s2, off);
    }
    if (l == 0) { sm[0] = s; sv[0] = s2; }
  }
  __syncthreads();
  float mean = sm[0] / (float)D_;
  float var  = sv[0] / (float)D_ - mean * mean;
  float inv  = rsqrtf(var + EPS_);
  float* y = Y + (size_t)row * D_;
  _Float16* y16 = Y16 + (size_t)row * D_;
  for (int j = threadIdx.x; j < D_; j += blockDim.x) {
    float v = (x[j] - mean) * inv * g[j] + be[j];
    y[j] = v;
    y16[j] = (_Float16)v;
  }
}

// ---------------- Mean-pool + final dense + sigmoid -------------------------

__global__ void pool_logit_kernel(const float* __restrict__ X,
                                  const float* __restrict__ Wf,
                                  const float* __restrict__ bf,
                                  float* __restrict__ out) {
  __shared__ float sp[8];
  int b = blockIdx.x;
  const float* x = X + (size_t)b * S_ * D_;
  float acc = 0.0f;
  for (int idx = threadIdx.x; idx < S_ * D_; idx += blockDim.x) {
    int d = idx & (D_ - 1);
    acc += x[idx] * Wf[d];
  }
#pragma unroll
  for (int off = 16; off > 0; off >>= 1) acc += __shfl_xor(acc, off);
  int w = threadIdx.x >> 5, l = threadIdx.x & 31;
  if (l == 0) sp[w] = acc;
  __syncthreads();
  if (threadIdx.x == 0) {
    float t = 0.0f;
#pragma unroll
    for (int i = 0; i < 8; ++i) t += sp[i];
    float logit = t / (float)S_ + bf[0];
    out[b]      = logit;
    out[B_ + b] = 1.0f / (1.0f + __expf(-logit));
  }
}

// ---------------------------------------------------------------------------
// Host orchestration
// ---------------------------------------------------------------------------

template <int TN>
static inline void launch_gemm(const _Float16* A, const _Float16* Wt, const float* bias,
                               const float* Rsd, float* C, _Float16* C16, _Float16* VT,
                               int M, int N, int K, int lda, int ldc, int relu,
                               int qcols, float qscale, hipStream_t stream) {
  dim3 grid(N / (16 * TN), M / 16);
  gemm_wmma_kernel<TN><<<grid, 32, 0, stream>>>(A, Wt, bias, Rsd, C, C16, VT,
                                                K, lda, ldc, relu, qcols, qscale);
}

static inline void launch_wt(const float* W, _Float16* Wt, int K, int N, hipStream_t stream) {
  wt_convert_kernel<<<(K * N + 255) / 256, 256, 0, stream>>>(W, Wt, K, N);
}

extern "C" void kernel_launch(void* const* d_in, const int* in_sizes, int n_in,
                              void* d_out, int out_size, void* d_ws, size_t ws_size,
                              hipStream_t stream) {
  const int*   tok = (const int*)  d_in[0];
  const float* emb = (const float*)d_in[1];
  const float* W0  = (const float*)d_in[2];
  const float* b0  = (const float*)d_in[3];
  const float* Wh  = (const float*)d_in[4];
  const float* bh  = (const float*)d_in[5];
  const float* Wo  = (const float*)d_in[6];
  const float* bo  = (const float*)d_in[7];
  const float* g1  = (const float*)d_in[8];
  const float* be1 = (const float*)d_in[9];
  const float* W1  = (const float*)d_in[10];
  const float* b1  = (const float*)d_in[11];
  const float* W2  = (const float*)d_in[12];
  const float* b2  = (const float*)d_in[13];
  const float* g2  = (const float*)d_in[14];
  const float* be2 = (const float*)d_in[15];
  const float* Wf  = (const float*)d_in[16];
  const float* bf  = (const float*)d_in[17];
  float* out = (float*)d_out;

  char* p = (char*)d_ws;
  auto alloc = [&](size_t bytes) {
    char* r = p;
    p += (bytes + 255) & ~(size_t)255;
    return r;
  };
  float*    x     = (float*)   alloc((size_t)BS_ * D_ * 4);
  float*    y     = (float*)   alloc((size_t)BS_ * D_ * 4);
  _Float16* x16   = (_Float16*)alloc((size_t)BS_ * D_ * 2);
  _Float16* hc16  = (_Float16*)alloc((size_t)BS_ * D_ * 2);        // head concat
  _Float16* qkv16 = (_Float16*)alloc((size_t)BS_ * TDH_ * 2);
  _Float16* vt    = (_Float16*)alloc((size_t)B_ * DH_ * S_ * 2);   // V^T per head
  _Float16* ffh16 = (_Float16*)alloc((size_t)FFN_CHUNK_ * FF_ * 2);
  _Float16* W0t   = (_Float16*)alloc((size_t)NB_ * TDH_ * D_ * 2);
  _Float16* Wht   = (_Float16*)alloc((size_t)NB_ * (H_-1) * TDH_ * DH_ * 2);
  _Float16* Wot   = (_Float16*)alloc((size_t)NB_ * D_ * D_ * 2);
  _Float16* W1t   = (_Float16*)alloc((size_t)NB_ * FF_ * D_ * 2);
  _Float16* W2t   = (_Float16*)alloc((size_t)NB_ * D_ * FF_ * 2);

  // ---- 0) per-launch weight transpose+convert to f16 [N,K] ----
  for (int l = 0; l < NB_; ++l) {
    launch_wt(W0 + (size_t)l * D_ * TDH_, W0t + (size_t)l * TDH_ * D_, D_, TDH_, stream);
    for (int i = 0; i < H_ - 1; ++i)
      launch_wt(Wh + ((size_t)l * (H_-1) + i) * DH_ * TDH_,
                Wht + ((size_t)l * (H_-1) + i) * TDH_ * DH_, DH_, TDH_, stream);
    launch_wt(Wo + (size_t)l * D_ * D_,  Wot + (size_t)l * D_ * D_,  D_, D_,  stream);
    launch_wt(W1 + (size_t)l * D_ * FF_, W1t + (size_t)l * FF_ * D_, D_, FF_, stream);
    launch_wt(W2 + (size_t)l * FF_ * D_, W2t + (size_t)l * D_ * FF_, FF_, D_, stream);
  }

  // ---- 1) embedding + positional encoding ----
  embed_pos_kernel<<<BS_, 256, 0, stream>>>(tok, emb, x, x16);

  const size_t shmem_attn = (size_t)QROWS_ * S_ * 4 + (size_t)S_ * DH_ * 2;  // 192 KB

  for (int l = 0; l < NB_; ++l) {
    // ---- attention: head 0 projection D -> 96, then 15 chained heads ----
    launch_gemm<2>(x16, W0t + (size_t)l * TDH_ * D_, b0 + (size_t)l * TDH_,
                   nullptr, nullptr, qkv16, vt, BS_, TDH_, D_, D_, TDH_, 0,
                   DH_, SCALE_, stream);
    attn_head_kernel<<<BS_ / QROWS_, 32, shmem_attn, stream>>>(qkv16, vt, hc16 + 0, D_);

    for (int i = 0; i < H_ - 1; ++i) {
      const _Float16* Whi = Wht + ((size_t)l * (H_-1) + i) * TDH_ * DH_;
      const float*    bhi = bh  + ((size_t)l * (H_-1) + i) * TDH_;
      launch_gemm<2>(hc16 + (size_t)i * DH_, Whi, bhi,
                     nullptr, nullptr, qkv16, vt, BS_, TDH_, DH_, D_, TDH_, 0,
                     DH_, SCALE_, stream);
      attn_head_kernel<<<BS_ / QROWS_, 32, shmem_attn, stream>>>(qkv16, vt,
                                                                 hc16 + (size_t)(i+1) * DH_, D_);
    }

    // ---- output projection + residual, then LN -> x/x16 ----
    launch_gemm<4>(hc16, Wot + (size_t)l * D_ * D_, bo + (size_t)l * D_,
                   /*Rsd=*/x, y, nullptr, nullptr, BS_, D_, D_, D_, D_, 0,
                   0, 1.0f, stream);
    ln_kernel<<<BS_, 256, 0, stream>>>(y, g1 + (size_t)l * D_, be1 + (size_t)l * D_, x, x16);

    // ---- FFN (chunked) + residual, then LN -> x/x16 ----
    for (int c = 0; c < BS_ / FFN_CHUNK_; ++c) {
      const float*    xr   = x   + (size_t)c * FFN_CHUNK_ * D_;
      const _Float16* xr16 = x16 + (size_t)c * FFN_CHUNK_ * D_;
      float*          yr   = y   + (size_t)c * FFN_CHUNK_ * D_;
      launch_gemm<4>(xr16, W1t + (size_t)l * FF_ * D_, b1 + (size_t)l * FF_,
                     nullptr, nullptr, ffh16, nullptr, FFN_CHUNK_, FF_, D_, D_, FF_,
                     /*relu=*/1, 0, 1.0f, stream);
      launch_gemm<4>(ffh16, W2t + (size_t)l * D_ * FF_, b2 + (size_t)l * D_,
                     /*Rsd=*/xr, yr, nullptr, nullptr, FFN_CHUNK_, D_, FF_, FF_, D_, 0,
                     0, 1.0f, stream);
    }
    ln_kernel<<<BS_, 256, 0, stream>>>(y, g2 + (size_t)l * D_, be2 + (size_t)l * D_, x, x16);
  }

  // ---- 2) mean-pool over S, final dense, sigmoid ----
  pool_logit_kernel<<<B_, 256, 0, stream>>>(x, Wf, bf, out);
}
